// SelfAttention2d_85134841742067
// MI455X (gfx1250) — matrix operand order repaired and checked by reading, not compiled, not run on hardware
//
#include <hip/hip_runtime.h>

typedef __bf16 bf16;
typedef __attribute__((ext_vector_type(8)))  __bf16 v8bf;
typedef __attribute__((ext_vector_type(16))) __bf16 v16bf;
typedef __attribute__((ext_vector_type(8)))  float  v8f;

#define C_DIM 512
#define N_DIM 4096
#define B_DIM 4
#define QK_SCALE 0.04419417382415922f   // 1/sqrt(512)

static __device__ __forceinline__ v16bf join8(v8bf a, v8bf b) {
  return __builtin_shufflevector(a, b, 0,1,2,3,4,5,6,7,8,9,10,11,12,13,14,15);
}
static __device__ __forceinline__ v8f wmma_bf16(v16bf a, v16bf b, v8f c) {
  return __builtin_amdgcn_wmma_f32_16x16x32_bf16(false, a, false, b, (short)0, c,
                                                 false, false);
}

// ---------------------------------------------------------------------------
// Kernel 0: LDS-tiled transpose+convert  x[b][c][n] (f32) -> xt[b][n][c] (bf16)
// ---------------------------------------------------------------------------
__global__ void __launch_bounds__(256) xpose_kernel(const float* __restrict__ x,
                                                    bf16* __restrict__ xt) {
  __shared__ float t[32][33];
  int blk = blockIdx.x;
  int b   = blk >> 11;           // 2048 blocks per batch (16 c-tiles * 128 n-tiles)
  int rem = blk & 2047;
  int ct  = rem >> 7;
  int nt  = rem & 127;
  int c0 = ct * 32, n0 = nt * 32;
  int tx = threadIdx.x & 31, ty = threadIdx.x >> 5;   // ty in 0..7
#pragma unroll
  for (int q = 0; q < 4; q++) {
    int cl = ty + q * 8;
    t[cl][tx] = x[((size_t)b * C_DIM + c0 + cl) * N_DIM + n0 + tx];
  }
  __syncthreads();
#pragma unroll
  for (int q = 0; q < 4; q++) {
    int nl = ty + q * 8;
    xt[((size_t)b * N_DIM + n0 + nl) * C_DIM + c0 + tx] = (bf16)t[tx][nl];
  }
}

// ---------------------------------------------------------------------------
// Kernel 1: fp32 -> bf16 weight convert
// ---------------------------------------------------------------------------
__global__ void cvt_kernel(const float* __restrict__ s, bf16* __restrict__ d, int n) {
  int i = blockIdx.x * blockDim.x + threadIdx.x;
  if (i < n) d[i] = (bf16)s[i];
}

// ---------------------------------------------------------------------------
// Kernel 2: QKV projections.  One 16x64 output strip per wave (4 accumulators)
// so the shared operand fragment feeds 4 WMMAs.
//   Q,K stored transposed [b][n][c] (scale+bias folded into Q),
//   V stored [b][c][n] so the attention B-operand is contiguous.
// ---------------------------------------------------------------------------
__global__ void __launch_bounds__(256) qkv_kernel(
    const bf16* __restrict__ xt,
    const bf16* __restrict__ wq, const bf16* __restrict__ wk,
    const bf16* __restrict__ wv,
    const float* __restrict__ bq, const float* __restrict__ bk,
    const float* __restrict__ bv,
    bf16* __restrict__ qt, bf16* __restrict__ kt, bf16* __restrict__ vm) {
  int w = threadIdx.x >> 5, lane = threadIdx.x & 31;
  int m16 = lane & 15, hi = lane >> 4;
  int g    = blockIdx.x * 8 + w;          // 24576 waves total
  int sel  = g >> 13;                      // 0=Q 1=K 2=V (8192 strips each)
  int rem  = g & 8191;
  int b    = rem >> 11;                    // 2048 strips per batch
  int rem2 = rem & 2047;
  int ti  = rem2 >> 3;                     // 256 i-tiles
  int tcg = rem2 & 7;                      // 8 strips of 64 output channels
  int i0 = ti * 16, co0 = tcg * 64;

  v8f acc[4] = {};
  if (sel < 2) {
    // D = (X^T tile)[i x c] * (W^T tiles)[c x o]   -> Q^T/K^T [n][c]
    const bf16* W    = sel ? wk : wq;
    const bf16* arow = xt + ((size_t)b * N_DIM + i0 + m16) * C_DIM;
    const bf16* brow0 = W + (size_t)(co0 +  0 + m16) * C_DIM;
    const bf16* brow1 = W + (size_t)(co0 + 16 + m16) * C_DIM;
    const bf16* brow2 = W + (size_t)(co0 + 32 + m16) * C_DIM;
    const bf16* brow3 = W + (size_t)(co0 + 48 + m16) * C_DIM;
#pragma unroll 2
    for (int kk = 0; kk < C_DIM; kk += 32) {
      v8bf  alo = *(const v8bf*)(arow + kk + hi * 16);
      v8bf  ahi = *(const v8bf*)(arow + kk + hi * 16 + 8);
      v16bf a   = join8(alo, ahi);
      acc[0] = wmma_bf16(a, *(const v16bf*)(brow0 + kk + hi * 16), acc[0]);
      acc[1] = wmma_bf16(a, *(const v16bf*)(brow1 + kk + hi * 16), acc[1]);
      acc[2] = wmma_bf16(a, *(const v16bf*)(brow2 + kk + hi * 16), acc[2]);
      acc[3] = wmma_bf16(a, *(const v16bf*)(brow3 + kk + hi * 16), acc[3]);
    }
    const float* bias = sel ? bk : bq;
    bf16* dst = sel ? kt : qt;
#pragma unroll
    for (int cg = 0; cg < 4; cg++) {
      float bb = bias[co0 + cg * 16 + m16];
#pragma unroll
      for (int r = 0; r < 8; r++) {
        float v = acc[cg][r] + bb;
        if (sel == 0) v *= QK_SCALE;
        dst[((size_t)b * N_DIM + i0 + r + hi * 8) * C_DIM + co0 + cg * 16 + m16] = (bf16)v;
      }
    }
  } else {
    // D = (Wv tiles)[o x c] * (X^T tile col-major)[c x i]   -> V [c][n]
    const bf16* arow0 = wv + (size_t)(co0 +  0 + m16) * C_DIM;
    const bf16* arow1 = wv + (size_t)(co0 + 16 + m16) * C_DIM;
    const bf16* arow2 = wv + (size_t)(co0 + 32 + m16) * C_DIM;
    const bf16* arow3 = wv + (size_t)(co0 + 48 + m16) * C_DIM;
    const bf16* brow  = xt + ((size_t)b * N_DIM + i0 + m16) * C_DIM;
#pragma unroll 2
    for (int kk = 0; kk < C_DIM; kk += 32) {
      v16bf bfrag = join8(*(const v8bf*)(brow + kk + hi * 8), *(const v8bf*)(brow + kk + 16 + hi * 8));
      acc[0] = wmma_bf16(join8(*(const v8bf*)(arow0 + kk + hi * 8),
                               *(const v8bf*)(arow0 + kk + 16 + hi * 8)), bfrag, acc[0]);
      acc[1] = wmma_bf16(join8(*(const v8bf*)(arow1 + kk + hi * 8),
                               *(const v8bf*)(arow1 + kk + 16 + hi * 8)), bfrag, acc[1]);
      acc[2] = wmma_bf16(join8(*(const v8bf*)(arow2 + kk + hi * 8),
                               *(const v8bf*)(arow2 + kk + 16 + hi * 8)), bfrag, acc[2]);
      acc[3] = wmma_bf16(join8(*(const v8bf*)(arow3 + kk + hi * 8),
                               *(const v8bf*)(arow3 + kk + 16 + hi * 8)), bfrag, acc[3]);
    }
#pragma unroll
    for (int cg = 0; cg < 4; cg++) {
#pragma unroll
      for (int r = 0; r < 8; r++) {
        int cout = co0 + cg * 16 + r + hi * 8;
        float v  = acc[cg][r] + bv[cout];
        vm[((size_t)b * C_DIM + cout) * N_DIM + i0 + m16] = (bf16)v;
      }
    }
  }
}

// ---------------------------------------------------------------------------
// Kernel 3: flash attention.  One block = one 32-query tile; 8 waves split the
// C=512 accumulator dim (64 channels each).  Partial scores reduced through a
// reused 16KB LDS buffer in two phases; wave0 (32 lanes = 32 rows) runs the
// online softmax; P is re-read in the 16-bit A-operand swizzle.
// ---------------------------------------------------------------------------
__global__ void __launch_bounds__(256) attn_kernel(const bf16* __restrict__ qt,
                                                   const bf16* __restrict__ kt,
                                                   const bf16* __restrict__ vm,
                                                   bf16* __restrict__ ot) {
  __shared__ bf16  Qs[32 * C_DIM];       // 32 KB query tile
  __shared__ float Sp[8][16 * 32];       // 16 KB per-wave partials (reused 2x)
  __shared__ float Sr[32 * 32];          //  4 KB reduced scores
  __shared__ bf16  Pt[32 * 32];          //  2 KB probabilities (bf16)
  __shared__ float rowM[32], rowL[32], rowF[32];

  int w = threadIdx.x >> 5, lane = threadIdx.x & 31;
  int m16 = lane & 15, hi = lane >> 4;
  int b  = blockIdx.x >> 7;              // 128 query tiles per batch
  int i0 = (blockIdx.x & 127) * 32;
  int c0 = w * 64;                       // this wave's channel slice

  {  // stage Q tile: 32 rows x 512 bf16 = 32 KB, rows contiguous in [n][c]
    const bf16* src = qt + ((size_t)b * N_DIM + i0) * C_DIM;
    int base = threadIdx.x * 64;
    *(v16bf*)&Qs[base]      = *(const v16bf*)&src[base];
    *(v16bf*)&Qs[base + 16] = *(const v16bf*)&src[base + 16];
    *(v16bf*)&Qs[base + 32] = *(const v16bf*)&src[base + 32];
    *(v16bf*)&Qs[base + 48] = *(const v16bf*)&src[base + 48];
  }
  if (threadIdx.x < 32) { rowM[threadIdx.x] = -1e30f; rowL[threadIdx.x] = 0.f; }
  __syncthreads();

  v8f o0[4] = {}, o1[4] = {};            // [iq==0/1][cg]  O accumulators

  for (int j0 = 0; j0 < N_DIM; j0 += 32) {
    if (j0 + 32 < N_DIM) {               // prefetch next K/V chunk (-> global_prefetch_b8)
      __builtin_prefetch(&kt[((size_t)b * N_DIM + j0 + 32 + m16) * C_DIM + c0], 0, 0);
      __builtin_prefetch(&vm[((size_t)b * C_DIM + c0 + m16) * N_DIM + j0 + 32], 0, 0);
    }

    // ---- partial S = Q^T K over this wave's 64-channel slice (4 tiles) ----
    v8f s00 = {}, s01 = {}, s10 = {}, s11 = {};
#pragma unroll
    for (int s = 0; s < 2; s++) {
      int cc = c0 + s * 32;
      v16bf a0 = join8(*(const v8bf*)&Qs[m16 * C_DIM + cc + hi * 8],
                       *(const v8bf*)&Qs[m16 * C_DIM + cc + 16 + hi * 8]);
      v16bf a1 = join8(*(const v8bf*)&Qs[(16 + m16) * C_DIM + cc + hi * 8],
                       *(const v8bf*)&Qs[(16 + m16) * C_DIM + cc + 16 + hi * 8]);
      v16bf b0 = join8(*(const v8bf*)&kt[((size_t)b * N_DIM + j0 + m16) * C_DIM + cc + hi * 8],
                       *(const v8bf*)&kt[((size_t)b * N_DIM + j0 + m16) * C_DIM + cc + 16 + hi * 8]);
      v16bf b1 = join8(*(const v8bf*)&kt[((size_t)b * N_DIM + j0 + 16 + m16) * C_DIM + cc + hi * 8],
                       *(const v8bf*)&kt[((size_t)b * N_DIM + j0 + 16 + m16) * C_DIM + cc + 16 + hi * 8]);
      s00 = wmma_bf16(a0, b0, s00);
      s01 = wmma_bf16(a0, b1, s01);
      s10 = wmma_bf16(a1, b0, s10);
      s11 = wmma_bf16(a1, b1, s11);
    }

    // ---- phase 1: reduce rows 0..15 across waves ----
#pragma unroll
    for (int r = 0; r < 8; r++) {
      Sp[w][(r + hi * 8) * 32 + m16]      = s00[r];
      Sp[w][(r + hi * 8) * 32 + 16 + m16] = s01[r];
    }
    __syncthreads();
    {
      int e = threadIdx.x * 2;
      float a0 = 0.f, a1 = 0.f;
#pragma unroll
      for (int ww = 0; ww < 8; ww++) { a0 += Sp[ww][e]; a1 += Sp[ww][e + 1]; }
      Sr[e] = a0; Sr[e + 1] = a1;
    }
    __syncthreads();

    // ---- phase 2: reduce rows 16..31 (reusing Sp) ----
#pragma unroll
    for (int r = 0; r < 8; r++) {
      Sp[w][(r + hi * 8) * 32 + m16]      = s10[r];
      Sp[w][(r + hi * 8) * 32 + 16 + m16] = s11[r];
    }
    __syncthreads();
    {
      int e = threadIdx.x * 2;
      float a0 = 0.f, a1 = 0.f;
#pragma unroll
      for (int ww = 0; ww < 8; ww++) { a0 += Sp[ww][e]; a1 += Sp[ww][e + 1]; }
      Sr[512 + e] = a0; Sr[512 + e + 1] = a1;
    }
    __syncthreads();

    // ---- online softmax: wave0, one lane per query row (32 rows) ----
    if (w == 0) {
      float mOld = rowM[lane];
      float mNew = mOld;
      for (int j = 0; j < 32; j++) mNew = fmaxf(mNew, Sr[lane * 32 + j]);
      float f = __expf(mOld - mNew);
      float l = rowL[lane] * f;
      for (int j = 0; j < 32; j++) {
        float p = __expf(Sr[lane * 32 + j] - mNew);
        l += p;
        Pt[lane * 32 + j] = (bf16)p;
      }
      rowM[lane] = mNew; rowL[lane] = l; rowF[lane] = f;
    }
    __syncthreads();

    // ---- rescale O, then O += P @ V^T (V fragments shared by both i-tiles) ----
    float fr0[8], fr1[8];
#pragma unroll
    for (int r = 0; r < 8; r++) { fr0[r] = rowF[r + hi * 8]; fr1[r] = rowF[16 + r + hi * 8]; }
    v16bf pa0 = join8(*(const v8bf*)&Pt[m16 * 32 + hi * 8],
                      *(const v8bf*)&Pt[m16 * 32 + 16 + hi * 8]);
    v16bf pa1 = join8(*(const v8bf*)&Pt[(16 + m16) * 32 + hi * 8],
                      *(const v8bf*)&Pt[(16 + m16) * 32 + 16 + hi * 8]);
    const size_t vb = (size_t)b * C_DIM;
#pragma unroll
    for (int cg = 0; cg < 4; cg++) {
#pragma unroll
      for (int r = 0; r < 8; r++) { o0[cg][r] *= fr0[r]; o1[cg][r] *= fr1[r]; }
      v16bf bv = join8(*(const v8bf*)&vm[(vb + c0 + cg * 16 + m16) * N_DIM + j0 + hi * 8],
                       *(const v8bf*)&vm[(vb + c0 + cg * 16 + m16) * N_DIM + j0 + 16 + hi * 8]);
      o0[cg] = wmma_bf16(pa0, bv, o0[cg]);
      o1[cg] = wmma_bf16(pa1, bv, o1[cg]);
    }
  }

  // ---- finalize: O /= l, store O^T [n][c] bf16 ----
#pragma unroll
  for (int cg = 0; cg < 4; cg++) {
#pragma unroll
    for (int r = 0; r < 8; r++) {
      {
        int row = r + hi * 8;
        float inv = 1.0f / rowL[row];
        bf16* dst = ot + ((size_t)b * N_DIM + i0 + row) * C_DIM;
        dst[c0 + cg * 16 + m16] = (bf16)(o0[cg][r] * inv);
      }
      {
        int row = 16 + r + hi * 8;
        float inv = 1.0f / rowL[row];
        bf16* dst = ot + ((size_t)b * N_DIM + i0 + row) * C_DIM;
        dst[c0 + cg * 16 + m16] = (bf16)(o1[cg][r] * inv);
      }
    }
  }
}

// ---------------------------------------------------------------------------
// Kernel 4: output projection + residual:  out = x + Wp @ O + bp   (fp32 out)
// 16x64 strip per wave: the O^T fragment feeds 4 WMMAs.
// ---------------------------------------------------------------------------
__global__ void __launch_bounds__(256) proj_kernel(const bf16* __restrict__ ot,
                                                   const bf16* __restrict__ wp,
                                                   const float* __restrict__ bp,
                                                   const float* __restrict__ x,
                                                   float* __restrict__ out) {
  int w = threadIdx.x >> 5, lane = threadIdx.x & 31;
  int m16 = lane & 15, hi = lane >> 4;
  int g   = blockIdx.x * 8 + w;          // 8192 waves total
  int b   = g >> 11;
  int rem = g & 2047;
  int tcg = rem >> 8, ti = rem & 255;
  int co0 = tcg * 64, i0 = ti * 16;

  const bf16* arow0 = wp + (size_t)(co0 +  0 + m16) * C_DIM;
  const bf16* arow1 = wp + (size_t)(co0 + 16 + m16) * C_DIM;
  const bf16* arow2 = wp + (size_t)(co0 + 32 + m16) * C_DIM;
  const bf16* arow3 = wp + (size_t)(co0 + 48 + m16) * C_DIM;
  const bf16* brow  = ot + ((size_t)b * N_DIM + i0 + m16) * C_DIM;
  v8f acc[4] = {};
#pragma unroll 2
  for (int kk = 0; kk < C_DIM; kk += 32) {
    v16bf bfrag = join8(*(const v8bf*)(brow + kk + hi * 8), *(const v8bf*)(brow + kk + 16 + hi * 8));
    acc[0] = wmma_bf16(join8(*(const v8bf*)(arow0 + kk + hi * 8),
                             *(const v8bf*)(arow0 + kk + 16 + hi * 8)), bfrag, acc[0]);
    acc[1] = wmma_bf16(join8(*(const v8bf*)(arow1 + kk + hi * 8),
                             *(const v8bf*)(arow1 + kk + 16 + hi * 8)), bfrag, acc[1]);
    acc[2] = wmma_bf16(join8(*(const v8bf*)(arow2 + kk + hi * 8),
                             *(const v8bf*)(arow2 + kk + 16 + hi * 8)), bfrag, acc[2]);
    acc[3] = wmma_bf16(join8(*(const v8bf*)(arow3 + kk + hi * 8),
                             *(const v8bf*)(arow3 + kk + 16 + hi * 8)), bfrag, acc[3]);
  }
#pragma unroll
  for (int cg = 0; cg < 4; cg++) {
#pragma unroll
    for (int r = 0; r < 8; r++) {
      int cout = co0 + cg * 16 + r + hi * 8;
      size_t idx = ((size_t)b * C_DIM + cout) * N_DIM + i0 + m16;
      out[idx] = x[idx] + acc[cg][r] + bp[cout];
    }
  }
}

// ---------------------------------------------------------------------------
extern "C" void kernel_launch(void* const* d_in, const int* in_sizes, int n_in,
                              void* d_out, int out_size, void* d_ws, size_t ws_size,
                              hipStream_t stream) {
  (void)in_sizes; (void)n_in; (void)out_size; (void)ws_size;
  const float* x  = (const float*)d_in[0];
  const float* wq = (const float*)d_in[1];
  const float* bq = (const float*)d_in[2];
  const float* wk = (const float*)d_in[3];
  const float* bk = (const float*)d_in[4];
  const float* wv = (const float*)d_in[5];
  const float* bv = (const float*)d_in[6];
  const float* wp = (const float*)d_in[7];
  const float* bp = (const float*)d_in[8];
  float* out = (float*)d_out;

  char* ws = (char*)d_ws;
  const size_t actBytes = (size_t)B_DIM * N_DIM * C_DIM * sizeof(bf16);  // 16 MB
  bf16* XT  = (bf16*)(ws);
  bf16* QT  = (bf16*)(ws + 1 * actBytes);
  bf16* KT  = (bf16*)(ws + 2 * actBytes);
  bf16* VM  = (bf16*)(ws + 3 * actBytes);
  bf16* OT  = (bf16*)(ws + 4 * actBytes);
  bf16* WQB = (bf16*)(ws + 5 * actBytes);
  bf16* WKB = WQB + (size_t)C_DIM * C_DIM;
  bf16* WVB = WKB + (size_t)C_DIM * C_DIM;
  bf16* WPB = WVB + (size_t)C_DIM * C_DIM;

  xpose_kernel<<<8192, 256, 0, stream>>>(x, XT);
  const int wn = C_DIM * C_DIM;
  cvt_kernel<<<wn / 256, 256, 0, stream>>>(wq, WQB, wn);
  cvt_kernel<<<wn / 256, 256, 0, stream>>>(wk, WKB, wn);
  cvt_kernel<<<wn / 256, 256, 0, stream>>>(wv, WVB, wn);
  cvt_kernel<<<wn / 256, 256, 0, stream>>>(wp, WPB, wn);
  qkv_kernel<<<3072, 256, 0, stream>>>(XT, WQB, WKB, WVB, bq, bk, bv, QT, KT, VM);
  attn_kernel<<<512, 256, 0, stream>>>(QT, KT, VM, OT);
  proj_kernel<<<1024, 256, 0, stream>>>(OT, WPB, bp, x, out);
}
